// SwinLayer_59425167508153
// MI455X (gfx1250) — matrix operand order repaired and checked by reading, not compiled, not run on hardware
//
#include <hip/hip_runtime.h>
#include <hip/hip_bf16.h>

typedef _Float16 h16;
typedef __attribute__((ext_vector_type(16))) _Float16 v16h;
typedef __attribute__((ext_vector_type(8)))  _Float16 v8h;
typedef __attribute__((ext_vector_type(8)))  float    v8f;

#define C_    64
#define SPTOT 409600          // 5*5*128*128
#define NTOT  26214400        // 64 * SPTOT
#define NWIN  6400            // 5*5*16*16 windows
#define C4_   256

__device__ __forceinline__ v8f vz8() {
  v8f r;
#pragma unroll
  for (int i = 0; i < 8; ++i) r[i] = 0.f;
  return r;
}

__device__ __forceinline__ v8f wmma_f16(v16h a, v16h b, v8f c) {
  return __builtin_amdgcn_wmma_f32_16x16x32_f16(false, a, false, b, (short)0, c,
                                                false, false);
}

// A-matrix 16x32 f16 fragment. base: row-major [row][k], ld elems, start col kb.
__device__ __forceinline__ v16h ldA(const h16* base, int ld, int kb, int lane) {
  int row = lane & 15;
  int ko  = (lane >> 4) * 8;
  const h16* p = base + row * ld + kb + ko;
  v8h lo = *(const v8h*)(p);
  v8h hi = *(const v8h*)(p + 16);
  v16h r;
#pragma unroll
  for (int i = 0; i < 8; ++i) { r[i] = lo[i]; r[i + 8] = hi[i]; }
  return r;
}

// B-matrix 32x16 f16 fragment. base: column-major [col][k] (k contiguous), ld elems.
__device__ __forceinline__ v16h ldB(const h16* base, int ld, int kb, int lane) {
  int col = lane & 15;
  int ko  = (lane >> 4) * 8;
  const h16* p = base + col * ld + kb + ko;
  v8h lo = *(const v8h*)(p);
  v8h hi = *(const v8h*)(p + 16);
  v16h r;
#pragma unroll
  for (int i = 0; i < 8; ++i) { r[i] = lo[i]; r[i + 8] = hi[i]; }
  return r;
}

// A fragment with real K=16, padded to 32 with zeros.
__device__ __forceinline__ v16h ldA16(const h16* base, int ld, int lane) {
  int row = lane & 15;
  int ko  = (lane >> 4) * 8;
  v8h lo = *(const v8h*)(base + row * ld + ko);
  v16h r;
#pragma unroll
  for (int i = 0; i < 8; ++i) { r[i] = lo[i]; r[i + 8] = (h16)0.f; }
  return r;
}

// B fragment with real K=16, padded to 32 with zeros (elements 8..15 hold K 16..31 -> 0).
__device__ __forceinline__ v16h ldB16(const h16* base, int ld, int lane) {
  v16h r;
#pragma unroll
  for (int i = 0; i < 16; ++i) r[i] = (h16)0.f;
  {
    const h16* p = base + (lane & 15) * ld + (lane >> 4) * 8;
    v8h lo = *(const v8h*)(p);
#pragma unroll
    for (int i = 0; i < 8; ++i) { r[i] = lo[i]; }
  }
  return r;
}

__device__ __forceinline__ float silu(float x) {
  return x * (1.f / (1.f + __expf(-x)));
}

// ---------------------------------------------------------------------------
// Prep: f16 weight copies + relative position bias table expansion
// ---------------------------------------------------------------------------
__global__ void k_prep(const float* __restrict__ qkv_w, const float* __restrict__ proj_w,
                       const float* __restrict__ mb_w1, const float* __restrict__ mb_w2,
                       const float* __restrict__ rpb, const int* __restrict__ rpi,
                       h16* wqkv, h16* wproj, h16* w1, h16* w2, float* biasb) {
  int i = blockIdx.x * 256 + threadIdx.x;
  if (i < 12288) wqkv[i] = (h16)qkv_w[i];
  if (i < 4096)  wproj[i] = (h16)proj_w[i];
  if (i < 16384) {
    w1[i] = (h16)mb_w1[i];
    w2[i] = (h16)mb_w2[i];
    int h  = i >> 12;          // head
    int ij = i & 4095;         // (i,j) in 64x64
    biasb[h * 4096 + ij] = rpb[rpi[ij] * 4 + h];
  }
}

// ---------------------------------------------------------------------------
// LayerNorm over C: writes f32 (C-major) and f16 (channel-last)
// ---------------------------------------------------------------------------
__global__ void k_ln1(const float* __restrict__ x, const float* __restrict__ w,
                      const float* __restrict__ b, float* __restrict__ t32,
                      h16* __restrict__ t16) {
  int sp = blockIdx.x * 256 + threadIdx.x;
  if (sp >= SPTOT) return;
  float mu = 0.f;
  for (int c = 0; c < C_; ++c) mu += x[(size_t)c * SPTOT + sp];
  mu *= (1.f / C_);
  float var = 0.f;
  for (int c = 0; c < C_; ++c) {
    float d = x[(size_t)c * SPTOT + sp] - mu;
    var += d * d;
  }
  var *= (1.f / C_);
  float inv = rsqrtf(var + 1e-5f);
  for (int c = 0; c < C_; ++c) {
    float v = (x[(size_t)c * SPTOT + sp] - mu) * inv * w[c] + b[c];
    t32[(size_t)c * SPTOT + sp] = v;
    t16[(size_t)sp * C_ + c]    = (h16)v;
  }
}

__global__ void k_ln2(const float* __restrict__ x1, const float* __restrict__ w,
                      const float* __restrict__ b, h16* __restrict__ t16) {
  int sp = blockIdx.x * 256 + threadIdx.x;
  if (sp >= SPTOT) return;
  float mu = 0.f;
  for (int c = 0; c < C_; ++c) mu += x1[(size_t)c * SPTOT + sp];
  mu *= (1.f / C_);
  float var = 0.f;
  for (int c = 0; c < C_; ++c) {
    float d = x1[(size_t)c * SPTOT + sp] - mu;
    var += d * d;
  }
  var *= (1.f / C_);
  float inv = rsqrtf(var + 1e-5f);
  for (int c = 0; c < C_; ++c)
    t16[(size_t)sp * C_ + c] =
        (h16)((x1[(size_t)c * SPTOT + sp] - mu) * inv * w[c] + b[c]);
}

// ---------------------------------------------------------------------------
// Shifted-window attention, 1 workgroup (4 waves = 4 heads) per 8x8 window.
// Writes x1 = x + attn_out (inverse shift on scatter).
// ---------------------------------------------------------------------------
__global__ __launch_bounds__(128) void k_attn(
    const h16* __restrict__ tf16, const float* __restrict__ x,
    const h16* __restrict__ wqkv, const float* __restrict__ qkvb,
    const h16* __restrict__ wproj, const float* __restrict__ projb,
    const float* __restrict__ biasb, float* __restrict__ x1) {
  extern __shared__ __align__(16) char smem_raw[];
  h16* xw = (h16*)smem_raw;      // [64][72]   window tokens, channel-contig
  h16* qk = xw + 64 * 72;        // [64][136]  [tok][0:64]=Q*scale  [64:128]=K
  h16* vt = qk + 64 * 136;       // [64][72]   rows = head*16+d, cols = tokens
  h16* sp = vt + 64 * 72;        // [4][64][72] S (f16), softmaxed in place
  h16* ol = sp + 4 * 64 * 72;    // [64][72]   attention output (heads concat)

  const int tid = threadIdx.x, lane = tid & 31, h = tid >> 5;  // wave == head
  const int wid = blockIdx.x;
  const int wb = wid & 15, hb = (wid >> 4) & 15;
  const int vv = (wid >> 8) % 5, uu = wid / 1280;
  const int imgbase = (uu * 5 + vv) * 16384;
  const int hi8 = (lane & 16) >> 1;  // +8 row offset for lanes 16..31
  const int ln15 = lane & 15;

  // ---- gather shifted window into LDS (roll +4: src = idx-4 mod 128) ----
  {
    int t = tid >> 1, half = tid & 1;
    int hh = t >> 3, ww = t & 7;
    int hf = (hb * 8 + hh - 4) & 127;
    int wf = (wb * 8 + ww - 4) & 127;
    const h16* src = tf16 + (size_t)(imgbase + hf * 128 + wf) * C_ + half * 32;
    h16* dst = xw + t * 72 + half * 32;
    *(v8h*)(dst)      = *(const v8h*)(src);
    *(v8h*)(dst + 8)  = *(const v8h*)(src + 8);
    *(v8h*)(dst + 16) = *(const v8h*)(src + 16);
    *(v8h*)(dst + 24) = *(const v8h*)(src + 24);
  }
  __syncthreads();

  // ---- QKV = Xw @ Wqkv^T : each wave computes its head's Q, K, V stripes ----
  // (three straight-line phases so the store target is compile-time constant)
  {
    const int ocol = h * 16;                       // Q columns
    float bias = qkvb[ocol + ln15];
#pragma unroll
    for (int mt = 0; mt < 4; ++mt) {
      v8f acc = vz8();
#pragma unroll
      for (int kb = 0; kb < 64; kb += 32)
        acc = wmma_f16(ldA(xw + mt * 16 * 72, 72, kb, lane),
                       ldB(wqkv + (size_t)ocol * 64, 64, kb, lane), acc);
#pragma unroll
      for (int r = 0; r < 8; ++r)
        qk[(mt * 16 + r + hi8) * 136 + h * 16 + ln15] =
            (h16)((acc[r] + bias) * 0.25f);
    }
  }
  {
    const int ocol = 64 + h * 16;                  // K columns
    float bias = qkvb[ocol + ln15];
#pragma unroll
    for (int mt = 0; mt < 4; ++mt) {
      v8f acc = vz8();
#pragma unroll
      for (int kb = 0; kb < 64; kb += 32)
        acc = wmma_f16(ldA(xw + mt * 16 * 72, 72, kb, lane),
                       ldB(wqkv + (size_t)ocol * 64, 64, kb, lane), acc);
#pragma unroll
      for (int r = 0; r < 8; ++r)
        qk[(mt * 16 + r + hi8) * 136 + 64 + h * 16 + ln15] = (h16)(acc[r] + bias);
    }
  }
  {
    const int ocol = 128 + h * 16;                 // V columns (stored transposed)
    float bias = qkvb[ocol + ln15];
#pragma unroll
    for (int mt = 0; mt < 4; ++mt) {
      v8f acc = vz8();
#pragma unroll
      for (int kb = 0; kb < 64; kb += 32)
        acc = wmma_f16(ldA(xw + mt * 16 * 72, 72, kb, lane),
                       ldB(wqkv + (size_t)ocol * 64, 64, kb, lane), acc);
#pragma unroll
      for (int r = 0; r < 8; ++r)
        vt[(h * 16 + ln15) * 72 + mt * 16 + r + hi8] = (h16)(acc[r] + bias);
    }
  }
  __syncthreads();

  // ---- S = Q K^T + bias (K padded 16->32), per head ----
#pragma unroll
  for (int mt = 0; mt < 4; ++mt)
#pragma unroll
    for (int nt = 0; nt < 4; ++nt) {
      v8f c;
      int ncol = nt * 16 + ln15;
#pragma unroll
      for (int r = 0; r < 8; ++r)
        c[r] = biasb[(h * 64 + mt * 16 + r + hi8) * 64 + ncol];
      v16h a = ldA16(qk + mt * 16 * 136 + h * 16, 136, lane);
      v16h b = ldB16(qk + nt * 16 * 136 + 64 + h * 16, 136, lane);
      v8f s = wmma_f16(a, b, c);
#pragma unroll
      for (int r = 0; r < 8; ++r)
        sp[(h * 64 + mt * 16 + r + hi8) * 72 + ncol] = (h16)s[r];
    }

  // ---- softmax in place: lane handles rows {lane, lane+32} of its head ----
#pragma unroll
  for (int rr = 0; rr < 2; ++rr) {
    h16* pr = sp + (h * 64 + lane + rr * 32) * 72;
    float mx = -1e30f;
    for (int j = 0; j < 64; ++j) mx = fmaxf(mx, (float)pr[j]);
    float sum = 0.f;
    for (int j = 0; j < 64; ++j) sum += __expf((float)pr[j] - mx);
    float inv = 1.f / sum;
    for (int j = 0; j < 64; ++j)
      pr[j] = (h16)(__expf((float)pr[j] - mx) * inv);
  }

  // ---- O = P @ V ----
#pragma unroll
  for (int mt = 0; mt < 4; ++mt) {
    v8f acc = vz8();
#pragma unroll
    for (int kb = 0; kb < 64; kb += 32)
      acc = wmma_f16(ldA(sp + (h * 64 + mt * 16) * 72, 72, kb, lane),
                     ldB(vt + (h * 16) * 72, 72, kb, lane), acc);
#pragma unroll
    for (int r = 0; r < 8; ++r)
      ol[(mt * 16 + r + hi8) * 72 + h * 16 + ln15] = (h16)acc[r];
  }
  __syncthreads();

  // ---- OUT = O @ Wproj^T, fused residual scatter with inverse shift ----
  {
    int nt = h;  // wave handles one 16-channel column tile
    int c = nt * 16 + ln15;
    float pb = projb[c];
#pragma unroll
    for (int mt = 0; mt < 4; ++mt) {
      v8f acc = vz8();
#pragma unroll
      for (int kb = 0; kb < 64; kb += 32)
        acc = wmma_f16(ldA(ol + mt * 16 * 72, 72, kb, lane),
                       ldB(wproj + (size_t)(nt * 16) * 64, 64, kb, lane), acc);
#pragma unroll
      for (int r = 0; r < 8; ++r) {
        int tok = mt * 16 + r + hi8;
        int hh = tok >> 3, ww = tok & 7;
        int hf = (hb * 8 + hh - 4) & 127;
        int wf = (wb * 8 + ww - 4) & 127;
        size_t gi = (size_t)c * SPTOT + imgbase + hf * 128 + wf;
        x1[gi] = x[gi] + acc[r] + pb;
      }
    }
  }
}

// ---------------------------------------------------------------------------
// MCA: axis pooling -> (64,133,133), 2x 1x1 MLP, 4 fused 1x1 convs, apply
// ---------------------------------------------------------------------------
__global__ void k_pool(const float* __restrict__ t, float* __restrict__ uvhw) {
  int idx = blockIdx.x * 256 + threadIdx.x;
  if (idx >= C_ * 17689) return;
  int c = idx / 17689, rem = idx % 17689;
  int row = rem / 133, col = rem % 133;
  const float* tc = t + (size_t)c * SPTOT;
  float s = 0.f;
  if (row < 128) {
    if (col < 128) {  // hw: mean over u,v
      for (int u = 0; u < 5; ++u)
        for (int v = 0; v < 5; ++v)
          s += tc[(u * 5 + v) * 16384 + row * 128 + col];
      s *= (1.f / 25.f);
    } else {          // hu: mean over v,w
      int u = col - 128;
      for (int v = 0; v < 5; ++v)
        for (int w = 0; w < 128; ++w)
          s += tc[(u * 5 + v) * 16384 + row * 128 + w];
      s *= (1.f / 640.f);
    }
  } else {
    int v = row - 128;
    if (col < 128) {  // vw: mean over u,h
      for (int u = 0; u < 5; ++u)
        for (int hh = 0; hh < 128; ++hh)
          s += tc[(u * 5 + v) * 16384 + hh * 128 + col];
      s *= (1.f / 640.f);
    } else {          // vu: mean over h,w
      int u = col - 128;
      for (int p = 0; p < 16384; ++p) s += tc[(u * 5 + v) * 16384 + p];
      s *= (1.f / 16384.f);
    }
  }
  uvhw[idx] = s;
}

__global__ void k_mca_mlp(const float* __restrict__ in, const float* __restrict__ w,
                          const float* __restrict__ b, float* __restrict__ out,
                          int do_silu) {
  int idx = blockIdx.x * 256 + threadIdx.x;
  if (idx >= C_ * 17689) return;
  int co = idx / 17689, pix = idx % 17689;
  float s = b[co];
  for (int ci = 0; ci < C_; ++ci) s += w[co * C_ + ci] * in[ci * 17689 + pix];
  out[idx] = do_silu ? silu(s) : s;
}

__global__ void k_mca_gates(const float* __restrict__ uvhw, const float* __restrict__ fw,
                            const float* __restrict__ fb, float* __restrict__ hwg,
                            float* __restrict__ uvg, float* __restrict__ uhg,
                            float* __restrict__ vwg) {
  int idx = blockIdx.x * 256 + threadIdx.x;
  if (idx >= C_ * 17689) return;
  if (idx < 1048576) {                       // hw gate (C,128,128)
    int c = idx / 16384, pix = idx % 16384;
    int hh = pix / 128, w = pix % 128;
    float s = fb[0 * C_ + c];
    for (int ci = 0; ci < C_; ++ci)
      s += fw[0 * 4096 + c * C_ + ci] * uvhw[ci * 17689 + hh * 133 + w];
    hwg[idx] = s;
  } else if (idx < 1048576 + 1600) {         // uv gate (C,5,5)
    int j = idx - 1048576;
    int c = j / 25, u = (j % 25) / 5, v = j % 5;
    float s = fb[1 * C_ + c];
    for (int ci = 0; ci < C_; ++ci)
      s += fw[1 * 4096 + c * C_ + ci] * uvhw[ci * 17689 + (128 + v) * 133 + 128 + u];
    uvg[j] = s;
  } else if (idx < 1048576 + 1600 + 40960) { // uh gate (C,5,128)
    int j = idx - 1048576 - 1600;
    int c = j / 640, u = (j % 640) / 128, hh = j % 128;
    float s = fb[2 * C_ + c];
    for (int ci = 0; ci < C_; ++ci)
      s += fw[2 * 4096 + c * C_ + ci] * uvhw[ci * 17689 + hh * 133 + 128 + u];
    uhg[j] = s;
  } else {                                   // vw gate (C,5,128)
    int j = idx - 1048576 - 1600 - 40960;
    int c = j / 640, v = (j % 640) / 128, w = j % 128;
    float s = fb[3 * C_ + c];
    for (int ci = 0; ci < C_; ++ci)
      s += fw[3 * 4096 + c * C_ + ci] * uvhw[ci * 17689 + (128 + v) * 133 + w];
    vwg[j] = s;
  }
}

__global__ void k_mca_apply(const float* __restrict__ t, const float* __restrict__ hwg,
                            const float* __restrict__ uvg, const float* __restrict__ uhg,
                            const float* __restrict__ vwg, float* __restrict__ x1) {
  size_t idx = (size_t)blockIdx.x * 256 + threadIdx.x;
  if (idx >= (size_t)NTOT) return;
  int c = (int)(idx / SPTOT);
  int sp = (int)(idx % SPTOT);
  int u = sp / 81920, r1 = sp % 81920;
  int v = r1 / 16384, pix = r1 % 16384;
  int hh = pix / 128, ww = pix % 128;
  float gate = hwg[c * 16384 + pix] + uvg[(c * 5 + u) * 5 + v] +
               uhg[(c * 5 + u) * 128 + hh] + vwg[(c * 5 + v) * 128 + ww];
  x1[idx] += t[idx] * gate;
}

// ---------------------------------------------------------------------------
// MBConv (per (u,v) image): expand WMMA GEMM, depthwise 5x5, project WMMA GEMM
// ---------------------------------------------------------------------------
__global__ __launch_bounds__(256) void k_expand(const h16* __restrict__ t2img,
                                                const h16* __restrict__ w1,
                                                const float* __restrict__ b1,
                                                h16* __restrict__ y1) {
  int lane = threadIdx.x & 31;
  int tile = blockIdx.x * 8 + (threadIdx.x >> 5);  // 16384 tiles
  int mt = tile & 15;      // 16 tiles over 256 out-channels
  int nt = tile >> 4;      // 1024 tiles over 16384 pixels
  v8f acc = vz8();
#pragma unroll
  for (int kb = 0; kb < 64; kb += 32)
    acc = wmma_f16(ldA(w1 + (size_t)mt * 16 * 64, 64, kb, lane),
                   ldB(t2img + (size_t)nt * 16 * 64, 64, kb, lane), acc);
  int pix = nt * 16 + (lane & 15);
  int hi8 = (lane & 16) >> 1;
#pragma unroll
  for (int r = 0; r < 8; ++r) {
    int co = mt * 16 + r + hi8;
    y1[(size_t)pix * C4_ + co] = (h16)silu(acc[r] + b1[co]);
  }
}

__global__ void k_dw(const h16* __restrict__ y1, const float* __restrict__ dw,
                     const float* __restrict__ db, h16* __restrict__ ydw) {
  int idx = blockIdx.x * 256 + threadIdx.x;  // 16384*256
  int co = idx & 255, pix = idx >> 8;
  int py = pix >> 7, px = pix & 127;
  float s = db[co];
#pragma unroll
  for (int dy = -2; dy <= 2; ++dy) {
    int yy = py + dy;
    if (yy < 0 || yy > 127) continue;
#pragma unroll
    for (int dx = -2; dx <= 2; ++dx) {
      int xx = px + dx;
      if (xx < 0 || xx > 127) continue;
      s += (float)y1[(size_t)(yy * 128 + xx) * C4_ + co] *
           dw[co * 25 + (dy + 2) * 5 + (dx + 2)];
    }
  }
  ydw[(size_t)pix * C4_ + co] = (h16)silu(s);
}

__global__ __launch_bounds__(256) void k_project(const h16* __restrict__ ydw,
                                                 const h16* __restrict__ w2,
                                                 const float* __restrict__ b2,
                                                 const float* __restrict__ x1,
                                                 float* __restrict__ out, int imgofs) {
  int lane = threadIdx.x & 31;
  int tile = blockIdx.x * 8 + (threadIdx.x >> 5);  // 4096 tiles
  int mt = tile & 3;       // 4 tiles over 64 out-channels
  int nt = tile >> 2;      // 1024 tiles over 16384 pixels
  v8f acc = vz8();
#pragma unroll
  for (int kb = 0; kb < 256; kb += 32)
    acc = wmma_f16(ldA(w2 + (size_t)mt * 16 * 256, 256, kb, lane),
                   ldB(ydw + (size_t)nt * 16 * 256, 256, kb, lane), acc);
  int pix = nt * 16 + (lane & 15);
  int hi8 = (lane & 16) >> 1;
#pragma unroll
  for (int r = 0; r < 8; ++r) {
    int co = mt * 16 + r + hi8;
    size_t gi = (size_t)co * SPTOT + imgofs + pix;
    out[gi] = acc[r] + b2[co] + x1[gi];
  }
}

// ---------------------------------------------------------------------------
extern "C" void kernel_launch(void* const* d_in, const int* in_sizes, int n_in,
                              void* d_out, int out_size, void* d_ws, size_t ws_size,
                              hipStream_t stream) {
  const float* x        = (const float*)d_in[0];
  const float* norm1_w  = (const float*)d_in[1];
  const float* norm1_b  = (const float*)d_in[2];
  const float* norm2_w  = (const float*)d_in[3];
  const float* norm2_b  = (const float*)d_in[4];
  const float* qkv_w    = (const float*)d_in[5];
  const float* qkv_b    = (const float*)d_in[6];
  const float* proj_w   = (const float*)d_in[7];
  const float* proj_b   = (const float*)d_in[8];
  const float* rpb      = (const float*)d_in[9];
  const int*   rpi      = (const int*)d_in[10];
  const float* mca_w1   = (const float*)d_in[11];
  const float* mca_b1   = (const float*)d_in[12];
  const float* mca_w2   = (const float*)d_in[13];
  const float* mca_b2   = (const float*)d_in[14];
  const float* mca_fw   = (const float*)d_in[15];
  const float* mca_fb   = (const float*)d_in[16];
  const float* mb_w1    = (const float*)d_in[17];
  const float* mb_b1    = (const float*)d_in[18];
  const float* mb_dw    = (const float*)d_in[19];
  const float* mb_db    = (const float*)d_in[20];
  const float* mb_w2    = (const float*)d_in[21];
  const float* mb_b2    = (const float*)d_in[22];
  float* out = (float*)d_out;

  char* base = (char*)d_ws;
  size_t off = 0;
  auto take = [&](size_t bytes) -> char* {
    char* r = base + off;
    off += (bytes + 255) & ~(size_t)255;
    return r;
  };
  h16*   t16    = (h16*)take((size_t)NTOT * 2);   // LN1 f16 (reused as LN2 f16)
  float* t32    = (float*)take((size_t)NTOT * 4); // LN1 f32 (MCA input)
  float* x1     = (float*)take((size_t)NTOT * 4); // x + attn (+ mca)
  h16*   wqkv16 = (h16*)take(12288 * 2);
  h16*   wproj16= (h16*)take(4096 * 2);
  h16*   w1_16  = (h16*)take(16384 * 2);
  h16*   w2_16  = (h16*)take(16384 * 2);
  float* biasb  = (float*)take(16384 * 4);
  float* uvhw0  = (float*)take((size_t)C_ * 17689 * 4);
  float* uvhw1  = (float*)take((size_t)C_ * 17689 * 4);
  float* hwg    = (float*)take(1048576 * 4);
  float* uvg    = (float*)take(1600 * 4);
  float* uhg    = (float*)take(40960 * 4);
  float* vwg    = (float*)take(40960 * 4);
  h16*   ybuf   = (h16*)take((size_t)16384 * C4_ * 2);
  h16*   ydwb   = (h16*)take((size_t)16384 * C4_ * 2);

  // weights + bias table
  k_prep<<<64, 256, 0, stream>>>(qkv_w, proj_w, mb_w1, mb_w2, rpb, rpi,
                                 wqkv16, wproj16, w1_16, w2_16, biasb);
  // LN1
  k_ln1<<<SPTOT / 256, 256, 0, stream>>>(x, norm1_w, norm1_b, t32, t16);
  // windowed attention -> x1 = x + attn
  k_attn<<<NWIN, 128, 81920, stream>>>(t16, x, wqkv16, qkv_b, wproj16, proj_b,
                                       biasb, x1);
  // MCA pipeline -> x1 += t * gates
  int gpix = (C_ * 17689 + 255) / 256;
  k_pool<<<gpix, 256, 0, stream>>>(t32, uvhw0);
  k_mca_mlp<<<gpix, 256, 0, stream>>>(uvhw0, mca_w1, mca_b1, uvhw1, 1);
  k_mca_mlp<<<gpix, 256, 0, stream>>>(uvhw1, mca_w2, mca_b2, uvhw0, 0);
  k_mca_gates<<<gpix, 256, 0, stream>>>(uvhw0, mca_fw, mca_fb, hwg, uvg, uhg, vwg);
  k_mca_apply<<<NTOT / 256, 256, 0, stream>>>(t32, hwg, uvg, uhg, vwg, x1);
  // LN2 -> channel-last f16 (per-image pixel-major)
  k_ln2<<<SPTOT / 256, 256, 0, stream>>>(x1, norm2_w, norm2_b, t16);
  // MBConv per (u,v) image, out = project(...) + x1
  for (int img = 0; img < 25; ++img) {
    const h16* t2img = t16 + (size_t)img * 16384 * C_;
    k_expand<<<2048, 256, 0, stream>>>(t2img, w1_16, mb_b1, ybuf);
    k_dw<<<16384, 256, 0, stream>>>(ybuf, mb_dw, mb_db, ydwb);
    k_project<<<512, 256, 0, stream>>>(ydwb, w2_16, mb_b2, x1, out, img * 16384);
  }
}
